// S3GNN_77386720739970
// MI455X (gfx1250) — matrix operand order repaired and checked by reading, not compiled, not run on hardware
//
#include <hip/hip_runtime.h>
#include <hip/hip_bf16.h>

// ---------------- problem constants (match reference) ----------------
constexpr int   Nn   = 50000;   // nodes
constexpr int   Ee   = 800000;  // edges per meta-path
constexpr int   Mm   = 3;       // meta-paths
constexpr int   INF_ = 256;     // in_size
constexpr int   NH   = 8;       // heads
constexpr int   HD   = 16;      // per-head dim
constexpr int   F    = 128;     // heads*dim
constexpr int   SAH  = 128;     // structure-attn hidden
constexpr int   OUTC = 8;       // classes
constexpr float SLOPE = 0.2f;

constexpr int   KT   = INF_ / 32;          // 8 K-tiles of 32
constexpr int   BFRAG_HALVES = 8 * 32 * 16; // one K-tile of packed B: 4096 halves (8 KB)

typedef __attribute__((ext_vector_type(16))) _Float16 v16h;
typedef __attribute__((ext_vector_type(8)))  _Float16 v8h;
typedef __attribute__((ext_vector_type(4)))  _Float16 v4h;
typedef __attribute__((ext_vector_type(8)))  float    v8f;

__device__ __forceinline__ float eluf(float x) {
    return x > 0.f ? x : (__expf(x) - 1.f);
}
__device__ __forceinline__ float lrelu(float x) {
    return x > 0.f ? x : SLOPE * x;
}
// float atomic max via sign-split integer atomics (works with -inf init = 0xFF800000)
__device__ __forceinline__ void atomicMaxF(float* addr, float v) {
    if (v >= 0.f) atomicMax((int*)addr, __float_as_int(v));
    else          atomicMin((unsigned int*)addr, __float_as_uint(v));
}

// ---------------- generic fill ----------------
__global__ void fill_kernel(unsigned* p, unsigned pat, long long n) {
    long long i = (long long)blockIdx.x * blockDim.x + threadIdx.x;
    long long stride = (long long)gridDim.x * blockDim.x;
    for (; i < n; i += stride) p[i] = pat;
}

__global__ void init_scalars_kernel(float* wmax, float* wsum) {
    *wmax = -__builtin_inff();
    *wsum = 0.f;
}

// ---------------- convert h (f32) -> hf16 (f16), vectorized x4 ----------------
__global__ void cvt_h_kernel(const float4* __restrict__ h4, v4h* __restrict__ o4,
                             long long n4) {
    long long i = (long long)blockIdx.x * blockDim.x + threadIdx.x;
    long long stride = (long long)gridDim.x * blockDim.x;
    for (; i < n4; i += stride) {
        float4 v = h4[i];
        v4h o;
        o[0] = (_Float16)v.x; o[1] = (_Float16)v.y;
        o[2] = (_Float16)v.z; o[3] = (_Float16)v.w;
        o4[i] = o;
    }
}

// ---------------- pack fc_w[m] into WMMA B-fragment order (f16) ----------------
// Fragment (kt, nt), lane l holds 16 contiguous halves:
//   b[i] = w[(kt*32 + ((l>=16)?16:0) + i)*128 + nt*16 + (l&15)]   i = 0..15
__global__ void pack_w_kernel(const float* __restrict__ w, _Float16* __restrict__ wpack) {
    int idx = blockIdx.x * blockDim.x + threadIdx.x;   // (kt*8 + nt)*32 + lane
    if (idx >= KT * 8 * 32) return;
    int lane = idx & 31;
    int nt   = (idx >> 5) & 7;
    int kt   = idx >> 8;
    int col  = nt * 16 + (lane & 15);
    int kb   = kt * 32 + ((lane >> 4) ? 16 : 0);
    _Float16* o = wpack + (long long)idx * 16;
    #pragma unroll
    for (int i = 0; i < 16; ++i)
        o[i] = (_Float16)w[(kb + i) * F + col];
}

// ---------------- WMMA GEMM: feat[N,128] = hf16[N,256] @ wpack ----------------
// 4 waves/block, each wave computes a 16-row x 128-col strip. Packed B K-tiles
// (8 KB) staged through LDS once per block; A fragments are two b128 loads.
__global__ void gemm_feat_kernel(const _Float16* __restrict__ hf,
                                 const _Float16* __restrict__ wpack,
                                 float* __restrict__ feat) {
    __shared__ __align__(16) _Float16 bsh[BFRAG_HALVES];   // 8 KB

    const int wave = threadIdx.x >> 5;
    const int lane = threadIdx.x & 31;
    const int row0 = (blockIdx.x * (blockDim.x >> 5) + wave) * 16;

    // tail waves must NOT return (they participate in barriers + WMMA);
    // clamp their loads, predicate their stores.
    int mrow = row0 + (lane & 15);
    if (mrow >= Nn) mrow = Nn - 1;
    const int khalf = (lane >> 4) * 16;         // lanes 16-31 start at K+16

    v8f acc[8];
    #pragma unroll
    for (int i = 0; i < 8; ++i)
        #pragma unroll
        for (int j = 0; j < 8; ++j) acc[i][j] = 0.f;

    for (int kt = 0; kt < KT; ++kt) {
        // cooperative copy of this K-tile's packed B into LDS (512 x uint4)
        const uint4* srcv = (const uint4*)(wpack + (long long)kt * BFRAG_HALVES);
        uint4* dstv = (uint4*)bsh;
        for (int i = threadIdx.x; i < BFRAG_HALVES / 8; i += blockDim.x)
            dstv[i] = srcv[i];
        __syncthreads();

        // A fragment: lane holds K = khalf+0..15 (contiguous f16)
        const _Float16* arow = hf + (long long)mrow * INF_ + kt * 32 + khalf;
        v8h alo = *(const v8h*)arow;          // 16B load
        v8h ahi = *(const v8h*)(arow + 8);    // 16B load
        v16h a = __builtin_shufflevector(alo, ahi,
                                         0, 1, 2, 3, 4, 5, 6, 7,
                                         8, 9, 10, 11, 12, 13, 14, 15);
        #pragma unroll
        for (int nt = 0; nt < 8; ++nt) {
            v16h b = *(const v16h*)&bsh[(nt * 32 + lane) * 16];  // ds_load_b128 x2
            acc[nt] = __builtin_amdgcn_wmma_f32_16x16x32_f16(
                false, a, false, b, (short)0, acc[nt], false, false);
        }
        __syncthreads();
    }

    // store D: VGPR j -> (M=j, N=lane) lanes0-15 ; (M=j+8, N=lane-16) lanes16-31
    const int colb = lane & 15;
    const int radd = (lane >> 4) * 8;
    #pragma unroll
    for (int nt = 0; nt < 8; ++nt)
        #pragma unroll
        for (int j = 0; j < 8; ++j) {
            int r = row0 + j + radd;
            if (r < Nn)
                feat[(long long)r * F + nt * 16 + colb] = acc[nt][j];
        }
}

// ---------------- per-node attention logits el/er ----------------
__global__ void elr_kernel(const float* __restrict__ feat,
                           const float* __restrict__ al,  // [8,16]
                           const float* __restrict__ ar,  // [8,16]
                           float* __restrict__ el, float* __restrict__ er) {
    int idx = blockIdx.x * blockDim.x + threadIdx.x;   // n*8 + head
    if (idx >= Nn * NH) return;
    int n = idx >> 3, hd = idx & 7;
    const float* f = feat + (long long)n * F + hd * HD;
    const float* a = al + hd * HD;
    const float* b = ar + hd * HD;
    float s1 = 0.f, s2 = 0.f;
    #pragma unroll
    for (int d = 0; d < HD; ++d) { float v = f[d]; s1 += v * a[d]; s2 += v * b[d]; }
    el[idx] = s1; er[idx] = s2;
}

// ---------------- edge softmax pass 1: segment max ----------------
__global__ void edge_max_kernel(const int* __restrict__ src, const int* __restrict__ dst,
                                const float* __restrict__ el, const float* __restrict__ er,
                                float* __restrict__ emax) {
    int e = blockIdx.x * blockDim.x + threadIdx.x;
    if (e >= Ee) return;
    int s = src[e], d = dst[e];
    #pragma unroll
    for (int hd = 0; hd < NH; ++hd) {
        float v = lrelu(el[s * NH + hd] + er[d * NH + hd]);
        atomicMaxF(&emax[d * NH + hd], v);
    }
}

// ---------------- edge softmax pass 2: segment sum of exp ----------------
__global__ void edge_sum_kernel(const int* __restrict__ src, const int* __restrict__ dst,
                                const float* __restrict__ el, const float* __restrict__ er,
                                const float* __restrict__ emax, float* __restrict__ denom) {
    int e = blockIdx.x * blockDim.x + threadIdx.x;
    if (e >= Ee) return;
    int s = src[e], d = dst[e];
    #pragma unroll
    for (int hd = 0; hd < NH; ++hd) {
        float v = lrelu(el[s * NH + hd] + er[d * NH + hd]);
        atomicAdd(&denom[d * NH + hd], __expf(v - emax[d * NH + hd]));
    }
}

// ---------------- edge pass 3: alpha-weighted message aggregation ----------------
__global__ void edge_agg_kernel(const int* __restrict__ src, const int* __restrict__ dst,
                                const float* __restrict__ el, const float* __restrict__ er,
                                const float* __restrict__ emax, const float* __restrict__ denom,
                                const float* __restrict__ feat, float* __restrict__ agg) {
    int e = blockIdx.x * blockDim.x + threadIdx.x;
    if (e >= Ee) return;
    int s = src[e], d = dst[e];
    #pragma unroll
    for (int hd = 0; hd < NH; ++hd) {
        float v = lrelu(el[s * NH + hd] + er[d * NH + hd]);
        float alpha = __expf(v - emax[d * NH + hd]) / denom[d * NH + hd];
        const float* fs = feat + (long long)s * F + hd * HD;
        float* ad = agg + (long long)d * F + hd * HD;
        #pragma unroll
        for (int dd = 0; dd < HD; ++dd)
            atomicAdd(&ad[dd], alpha * fs[dd]);
    }
}

// ---------------- structure-attention logit per node ----------------
// wl[n] = tanh(elu(agg[n,:]) @ sa_w1 + b1) @ sa_w2   (one 128-thread block per node)
__global__ void wlogit_kernel(const float* __restrict__ agg,
                              const float* __restrict__ w1,   // [128,128]
                              const float* __restrict__ b1,   // [128]
                              const float* __restrict__ w2,   // [128]
                              float* __restrict__ wl) {
    __shared__ float zs[SAH];
    __shared__ float red[SAH];
    const int n = blockIdx.x;
    const int t = threadIdx.x;
    zs[t] = eluf(agg[(long long)n * F + t]);
    __syncthreads();
    float acc = b1[t];
    for (int k = 0; k < F; ++k) acc += zs[k] * w1[k * SAH + t];
    red[t] = tanhf(acc) * w2[t];
    __syncthreads();
    for (int off = SAH / 2; off > 0; off >>= 1) {
        if (t < off) red[t] += red[t + off];
        __syncthreads();
    }
    if (t == 0) wl[n] = red[0];
}

// ---------------- softmax-over-nodes reductions ----------------
__global__ void reduce_max_kernel(const float* __restrict__ wl, float* __restrict__ wmax) {
    __shared__ float red[256];
    int t = threadIdx.x;
    float m = -3.402823466e38f;
    for (long long i = (long long)blockIdx.x * blockDim.x + t; i < Nn;
         i += (long long)gridDim.x * blockDim.x)
        m = fmaxf(m, wl[i]);
    red[t] = m; __syncthreads();
    for (int off = 128; off > 0; off >>= 1) {
        if (t < off) red[t] = fmaxf(red[t], red[t + off]);
        __syncthreads();
    }
    if (t == 0) atomicMaxF(wmax, red[0]);
}

__global__ void reduce_sumexp_kernel(const float* __restrict__ wl,
                                     const float* __restrict__ wmax,
                                     float* __restrict__ wsum) {
    __shared__ float red[256];
    int t = threadIdx.x;
    float mx = wmax[0];
    float s = 0.f;
    for (long long i = (long long)blockIdx.x * blockDim.x + t; i < Nn;
         i += (long long)gridDim.x * blockDim.x)
        s += __expf(wl[i] - mx);
    red[t] = s; __syncthreads();
    for (int off = 128; off > 0; off >>= 1) {
        if (t < off) red[t] += red[t + off];
        __syncthreads();
    }
    if (t == 0) atomicAdd(wsum, red[0]);
}

// ---------------- fused[m,:] = sum_n beta[n] * elu(agg[n,:]) ----------------
__global__ void fused_accum_kernel(const float* __restrict__ agg,
                                   const float* __restrict__ wl,
                                   const float* __restrict__ wmax,
                                   const float* __restrict__ wsum,
                                   float* __restrict__ fused) {   // points at fused + m*128
    const int t = threadIdx.x;                    // 128 threads, one per dim
    const float mx = wmax[0], sm = wsum[0];
    int n0 = blockIdx.x * 256;
    int n1 = n0 + 256; if (n1 > Nn) n1 = Nn;
    float acc = 0.f;
    for (int n = n0; n < n1; ++n) {
        float beta = __expf(wl[n] - mx) / sm;
        acc += beta * eluf(agg[(long long)n * F + t]);
    }
    atomicAdd(&fused[t], acc);
}

// ---------------- final projection: out[m,o] = fused[m]·pred_w[:,o] + pred_b[o] ----------------
__global__ void predict_kernel(const float* __restrict__ fused,
                               const float* __restrict__ pw,  // [128,8]
                               const float* __restrict__ pb,  // [8]
                               float* __restrict__ out) {     // [3,8]
    int idx = threadIdx.x;
    if (idx >= Mm * OUTC) return;
    int m = idx >> 3, o = idx & 7;
    float s = pb[o];
    for (int k = 0; k < F; ++k) s += fused[m * F + k] * pw[k * OUTC + o];
    out[idx] = s;
}

// ---------------- host orchestration ----------------
extern "C" void kernel_launch(void* const* d_in, const int* in_sizes, int n_in,
                              void* d_out, int out_size, void* d_ws, size_t ws_size,
                              hipStream_t stream) {
    const float* h      = (const float*)d_in[0];
    const int*   edges  = (const int*)  d_in[1];   // [M,2,E]
    const float* fc_w   = (const float*)d_in[2];   // [M,256,128]
    const float* attn_l = (const float*)d_in[3];   // [M,8,16]
    const float* attn_r = (const float*)d_in[4];   // [M,8,16]
    const float* sa_w1  = (const float*)d_in[5];   // [128,128]
    const float* sa_b1  = (const float*)d_in[6];   // [128]
    const float* sa_w2  = (const float*)d_in[7];   // [128,1]
    const float* pred_w = (const float*)d_in[8];   // [128,8]
    const float* pred_b = (const float*)d_in[9];   // [8]
    float* out = (float*)d_out;

    // workspace carve (~84 MB); meta-paths processed sequentially so the hot
    // feat/agg buffers (25.6 MB each) stay resident in the 192 MB L2.
    _Float16* hf16  = (_Float16*)d_ws;                         // Nn*256 halves
    _Float16* wpack = hf16 + (long long)Nn * INF_;             // KT*8*32*16 halves
    float* feat  = (float*)(wpack + KT * 8 * 32 * 16);         // Nn*128
    float* agg   = feat  + (long long)Nn * F;                  // Nn*128
    float* el    = agg   + (long long)Nn * F;                  // Nn*8
    float* er    = el    + Nn * NH;                            // Nn*8
    float* emax  = er    + Nn * NH;                            // Nn*8
    float* denom = emax  + Nn * NH;                            // Nn*8
    float* wl    = denom + Nn * NH;                            // Nn
    float* fused = wl    + Nn;                                 // 3*128
    float* wmax  = fused + Mm * F;                             // 1
    float* wsum  = wmax  + 1;                                  // 1

    const unsigned NEGINF = 0xFF800000u;

    // one-time per launch: convert h to f16, zero the fused accumulator
    cvt_h_kernel<<<2048, 256, 0, stream>>>((const float4*)h, (v4h*)hf16,
                                           (long long)Nn * INF_ / 4);
    fill_kernel<<<4, 256, 0, stream>>>((unsigned*)fused, 0u, Mm * F);

    const int edgeBlocks = (Ee + 255) / 256;
    const int gemmBlocks = ((Nn / 16) + 3) / 4;        // 4 waves/block, 16 rows/wave

    for (int m = 0; m < Mm; ++m) {
        const int*   src = edges  + (long long)m * 2 * Ee;
        const int*   dst = src + Ee;
        const float* wm  = fc_w   + (long long)m * INF_ * F;
        const float* alm = attn_l + (long long)m * NH * HD;
        const float* arm = attn_r + (long long)m * NH * HD;

        // 1) pack weights into WMMA B-fragment order, then WMMA GEMM
        pack_w_kernel<<<(KT * 8 * 32 + 255) / 256, 256, 0, stream>>>(wm, wpack);
        gemm_feat_kernel<<<gemmBlocks, 128, 0, stream>>>(hf16, wpack, feat);

        // 2) per-node attention logits
        elr_kernel<<<(Nn * NH + 255) / 256, 256, 0, stream>>>(feat, alm, arm, el, er);

        // 3) init segment buffers
        fill_kernel<<<512, 256, 0, stream>>>((unsigned*)emax,  NEGINF, (long long)Nn * NH);
        fill_kernel<<<512, 256, 0, stream>>>((unsigned*)denom, 0u,     (long long)Nn * NH);
        fill_kernel<<<2048, 256, 0, stream>>>((unsigned*)agg,  0u,     (long long)Nn * F);
        init_scalars_kernel<<<1, 1, 0, stream>>>(wmax, wsum);

        // 4) edge softmax + aggregation (L2-resident atomics)
        edge_max_kernel<<<edgeBlocks, 256, 0, stream>>>(src, dst, el, er, emax);
        edge_sum_kernel<<<edgeBlocks, 256, 0, stream>>>(src, dst, el, er, emax, denom);
        edge_agg_kernel<<<edgeBlocks, 256, 0, stream>>>(src, dst, el, er, emax, denom, feat, agg);

        // 5) structure-attention logit per node
        wlogit_kernel<<<Nn, SAH, 0, stream>>>(agg, sa_w1, sa_b1, sa_w2, wl);

        // 6) softmax over nodes, then weighted node-sum into fused[m,:]
        reduce_max_kernel<<<512, 256, 0, stream>>>(wl, wmax);
        reduce_sumexp_kernel<<<512, 256, 0, stream>>>(wl, wmax, wsum);
        fused_accum_kernel<<<(Nn + 255) / 256, F, 0, stream>>>(agg, wl, wmax, wsum,
                                                               fused + m * F);
    }

    // 7) final tiny projection -> out[3,8]
    predict_kernel<<<1, 32, 0, stream>>>(fused, pred_w, pred_b, out);
}
